// ScaledDotProductAttention_28372553958027
// MI455X (gfx1250) — matrix operand order repaired and checked by reading, not compiled, not run on hardware
//
#include <hip/hip_runtime.h>
#include <math.h>

typedef _Float16 h16;
typedef __attribute__((ext_vector_type(16))) _Float16 v16h;
typedef __attribute__((ext_vector_type(8)))  _Float16 v8h;
typedef __attribute__((ext_vector_type(4)))  _Float16 v4h;
typedef __attribute__((ext_vector_type(2)))  _Float16 v2h;
typedef __attribute__((ext_vector_type(8)))  float    v8f;

static constexpr int BH = 128, S = 1024, D = 64;

// int8 symmetric fake-quant forward: clamp(rndne(x*inv_s), -128, 127) * s
__device__ __forceinline__ float fq(float x, float inv_s, float s) {
    return fminf(fmaxf(rintf(x * inv_s), -128.f), 127.f) * s;
}

__device__ __forceinline__ v16h cat16(v8h lo, v8h hi) {
    return __builtin_shufflevector(lo, hi, 0,1,2,3,4,5,6,7,8,9,10,11,12,13,14,15);
}

union HU { v8h v; v2h p[4]; };

__global__ __launch_bounds__(128)
void ScaledDotProductAttention_28372553958027_kernel(
    const float* __restrict__ q, const float* __restrict__ k,
    const float* __restrict__ v,
    const float* __restrict__ qk_scale, const float* __restrict__ a_scale,
    const float* __restrict__ av_scale, float* __restrict__ out)
{
    // 152 KB LDS (of 320 KB / WGP) -> 2 workgroups resident per WGP (8 waves)
    __shared__ alignas(16) _Float16 sQ [64 * 64];      //  8 KB  64 query rows, row-major f16
    __shared__ alignas(16) _Float16 sK [64 * 64];      //  8 KB  64-key chunk, row-major
    __shared__ alignas(16) _Float16 sVt[64 * 64];      //  8 KB  V chunk transposed: [d][key]
    __shared__ alignas(16) _Float16 sS [4 * 16 * 1024];// 128 KB scores->exp->probs, per-wave slice

    const int tid  = threadIdx.x;
    const int lane = tid & 31;            // wave32
    const int wave = tid >> 5;            // 4 waves, one 16-row query tile each
    const int bh   = blockIdx.x >> 4;     // / (S/64)
    const int qgrp = blockIdx.x & 15;     // 64-row query group
    const int row  = lane & 15;           // M (A,C) or N (B) index
    const int half = lane >> 4;           // lane group 0/1

    const float qk_s = qk_scale[0], a_s = a_scale[0], av_s = av_scale[0];
    const float qk_i = 1.0f / qk_s, a_i = 1.0f / a_s, av_i = 1.0f / av_s;
    const float s1 = 0.125f * qk_i;       // fold 1/sqrt(D) into quant reciprocal

    _Float16* sSw = &sS[wave * 16 * 1024];

    // ---------------- stage 64 query rows fp32 -> f16 (cooperative) ----------------
    {
        const float4* gq = (const float4*)(q + ((size_t)bh * S + (size_t)qgrp * 64) * D);
        for (int j = tid; j < 64 * 64 / 4; j += 128) {
            float4 f = gq[j];
            v4h hh = { (h16)f.x, (h16)f.y, (h16)f.z, (h16)f.w };
            *(v4h*)&sQ[j * 4] = hh;
        }
    }
    __syncthreads();

    // A-fragments of this wave's Q tile. 16-bit A 16x32 layout:
    // lanes 0-15 K-base 0, lanes 16-31 K-base 8; elems 0..7 = base..+7, 8..15 = base+16..+23.
    v16h aQ[2];
    {
        const int bk = half * 8;
        #pragma unroll
        for (int kc = 0; kc < 2; ++kc) {
            v8h lo = *(const v8h*)&sQ[(wave * 16 + row) * 64 + kc * 32 + bk];
            v8h hi = *(const v8h*)&sQ[(wave * 16 + row) * 64 + kc * 32 + bk + 16];
            aQ[kc] = cat16(lo, hi);
        }
    }

    // ---------------- phase 2: S = fq(QK^T / sqrt(D)) ----------------
    const float* gk_base = k + (size_t)bh * S * D;
    for (int tc = 0; tc < S / 64; ++tc) {
        const float4* gk = (const float4*)(gk_base + (size_t)tc * 64 * D);
        for (int j = tid; j < 64 * 64 / 4; j += 128) {
            float4 f = gk[j];
            v4h hh = { (h16)f.x, (h16)f.y, (h16)f.z, (h16)f.w };
            *(v4h*)&sK[j * 4] = hh;
        }
        __syncthreads();

        #pragma unroll
        for (int sub = 0; sub < 4; ++sub) {
            // B = K^T (32x16): element (kd, n) = sK[(sub*16+n)*64 + kd]; contiguous in kd.
            v8f acc = {};
            #pragma unroll
            for (int kc = 0; kc < 2; ++kc) {
                const int base = (sub * 16 + row) * 64 + kc * 32 + half * 8;
                v8h lo = *(const v8h*)&sK[base];
                v8h hi = *(const v8h*)&sK[base + 16];
                v16h b = cat16(lo, hi);
                acc = __builtin_amdgcn_wmma_f32_16x16x32_f16(
                    false, aQ[kc], false, b, (short)0, acc, false, false);
            }
            // C/D layout: VGPR r, lane L -> (M = r + 8*(L>>4), N = L&15)
            #pragma unroll
            for (int r = 0; r < 8; ++r) {
                float val = fq(acc[r], s1, qk_s);
                sSw[(r + 8 * half) * 1024 + (tc * 4 + sub) * 16 + row] = (h16)val;
            }
        }
        __syncthreads();
    }

    // ---------------- phase 3: softmax + fake-quant probs (packed f16, in place) ---------
    {
        const int base = row * 1024 + half * 512;    // 2 lanes per row, 512 cols each
        v2h m2 = { (h16)-65504.f, (h16)-65504.f };
        for (int j = 0; j < 64; ++j) {
            HU u; u.v = *(const v8h*)&sSw[base + j * 8];
            #pragma unroll
            for (int c = 0; c < 4; ++c) m2 = __builtin_elementwise_max(m2, u.p[c]);
        }
        float m = fmaxf((float)m2[0], (float)m2[1]);
        m = fmaxf(m, __shfl_xor(m, 16, 32));

        const v2h mm  = { (h16)m, (h16)m };
        const v2h l2e = { (h16)1.44269504f, (h16)1.44269504f };
        v2h sum2 = { (h16)0.f, (h16)0.f };
        for (int j = 0; j < 64; ++j) {
            HU u; u.v = *(const v8h*)&sSw[base + j * 8];
            #pragma unroll
            for (int c = 0; c < 4; ++c) {
                v2h t = (u.p[c] - mm) * l2e;                 // v_pk_sub/mul_f16
                v2h e = __builtin_elementwise_exp2(t);       // native v_exp_f16 x2
                sum2 += e;
                u.p[c] = e;
            }
            *(v8h*)&sSw[base + j * 8] = u.v;
        }
        float sum = (float)sum2[0] + (float)sum2[1];
        sum += __shfl_xor(sum, 16, 32);
        const float inv = 1.f / sum;

        const float c1f = inv * a_i;                          // fold 1/sum and 1/a_s
        const v2h c1   = { (h16)c1f, (h16)c1f };
        const v2h cs   = { (h16)a_s, (h16)a_s };
        const v2h q127 = { (h16)127.f, (h16)127.f };
        for (int j = 0; j < 64; ++j) {
            HU u; u.v = *(const v8h*)&sSw[base + j * 8];
            #pragma unroll
            for (int c = 0; c < 4; ++c) {
                v2h pq = u.p[c] * c1;                         // e*inv/a_s in [0,127]
                v2h qq = __builtin_elementwise_rint(pq);      // v_rndne_f16
                qq = __builtin_elementwise_min(qq, q127);
                u.p[c] = qq * cs;                             // back to prob grid
            }
            *(v8h*)&sSw[base + j * 8] = u.v;
        }
    }
    __syncthreads();

    // ---------------- phase 4: O = A @ V ----------------
    v8f o[4] = { {}, {}, {}, {} };
    const float* gv_base = v + (size_t)bh * S * D;
    for (int tc = 0; tc < S / 64; ++tc) {
        // stage V rows tc*64..+63 transposed: sVt[d*64 + key]  (cooperative)
        const float4* gv = (const float4*)(gv_base + (size_t)tc * 64 * D);
        for (int j = tid; j < 64 * 64 / 4; j += 128) {
            float4 f = gv[j];
            int kk = j >> 4, d4 = (j & 15) * 4;
            sVt[(d4 + 0) * 64 + kk] = (h16)f.x;
            sVt[(d4 + 1) * 64 + kk] = (h16)f.y;
            sVt[(d4 + 2) * 64 + kk] = (h16)f.z;
            sVt[(d4 + 3) * 64 + kk] = (h16)f.w;
        }
        __syncthreads();

        #pragma unroll
        for (int sub = 0; sub < 2; ++sub) {
            const int ko = tc * 64 + sub * 32;
            // A fragment from quantized probs
            const int abk = half * 8;
            v8h alo = *(const v8h*)&sSw[row * 1024 + ko + abk];
            v8h ahi = *(const v8h*)&sSw[row * 1024 + ko + abk + 16];
            v16h a = cat16(alo, ahi);

            #pragma unroll
            for (int nt = 0; nt < 4; ++nt) {
                const int bb = (nt * 16 + row) * 64 + sub * 32 + half * 8;
                v8h blo = *(const v8h*)&sVt[bb];
                v8h bhi = *(const v8h*)&sVt[bb + 16];
                v16h b = cat16(blo, bhi);
                o[nt] = __builtin_amdgcn_wmma_f32_16x16x32_f16(
                    false, a, false, b, (short)0, o[nt], false, false);
            }
        }
        __syncthreads();
    }

    // ---------------- store: fq(O, av_scale) ----------------
    float* go = out + ((size_t)bh * S + (size_t)qgrp * 64 + (size_t)wave * 16) * D;
    #pragma unroll
    for (int nt = 0; nt < 4; ++nt) {
        #pragma unroll
        for (int r = 0; r < 8; ++r) {
            go[(r + 8 * half) * 64 + nt * 16 + row] = fq(o[nt][r], av_i, av_s);
        }
    }
}

extern "C" void kernel_launch(void* const* d_in, const int* in_sizes, int n_in,
                              void* d_out, int out_size, void* d_ws, size_t ws_size,
                              hipStream_t stream) {
    const float* q        = (const float*)d_in[0];
    const float* k        = (const float*)d_in[1];
    const float* v        = (const float*)d_in[2];
    const float* qk_scale = (const float*)d_in[3];
    const float* a_scale  = (const float*)d_in[4];
    const float* av_scale = (const float*)d_in[5];
    float* out = (float*)d_out;

    dim3 grid(BH * (S / 64));   // 2048 workgroups: 4 waves, 64 query rows each
    dim3 block(128);
    ScaledDotProductAttention_28372553958027_kernel<<<grid, block, 0, stream>>>(
        q, k, v, qk_scale, a_scale, av_scale, out);
}
